// DIN_66314295050867
// MI455X (gfx1250) — compile-verified, matrix operand order checked by reading, not run on hardware
//
#include <hip/hip_runtime.h>
#include <hip/hip_bf16.h>
#include <math.h>

#define BB 1024
#define LL 200
#define DD 64
#define BL (BB*LL)          /* 204800 rows */
#define NTILES (BL/16)      /* 12800 16-row WMMA tiles */

typedef __bf16 bf16;
typedef __attribute__((ext_vector_type(4)))  __bf16 v4bf;
typedef __attribute__((ext_vector_type(8)))  __bf16 v8bf;
typedef __attribute__((ext_vector_type(16))) __bf16 v16bf;
typedef __attribute__((ext_vector_type(8)))  float  v8f;

// native cast -> backend can use v_cvt_pk_bf16_f32
static __device__ __forceinline__ bf16 f2bf(float f) { return (bf16)f; }

static __device__ __forceinline__ float dice_f(float x, float mu, float rs, float al) {
  float p = 1.0f / (1.0f + expf(-(x - mu) * rs));
  return p * x + al * (1.0f - p) * x;
}

// ---------------- K0: zero stats + loss slot ----------------
__global__ void din_init(float* st, float* out) {
  int t = threadIdx.x;
  if (t < 96) st[t] = 0.0f;
  if (t == 0) out[0] = 0.0f;
}

// ---------------- K0b: pre-swizzle aw0 into WMMA B-fragment order (bf16) ----
// frag f = kb*2+nt (32 frags). Element layout per lane matches the 16-bit
// B-matrix VGPR layout: halves 0-7 = K koff..koff+7, 8-15 = K koff+16..+23,
// col n = nt*16 + (lane&15), koff = (lane>>4)*8.
__global__ void din_cvtw(const float* __restrict__ aw0, bf16* __restrict__ wbf) {
  int t = blockIdx.x * blockDim.x + threadIdx.x;   // 0..1023
  if (t >= 32 * 32) return;
  int f = t >> 5, lane = t & 31;
  int kb = f >> 1, nt = f & 1;
  int n = nt * 16 + (lane & 15);
  int koff = (lane >> 4) * 8;
  bf16* dst = wbf + (size_t)t * 16;
  const float* src = aw0 + (size_t)n * 512 + kb * 32 + koff;
#pragma unroll
  for (int i = 0; i < 8; ++i) {
    dst[i]     = f2bf(src[i]);
    dst[8 + i] = f2bf(src[16 + i]);
  }
}

// ---------------- K1: gather + x-form + GEMM0 (512->32) + stats ----------------
__global__ __launch_bounds__(64) void din_act0(
    const int* __restrict__ hist_i, const int* __restrict__ hist_c,
    const int* __restrict__ tgt_i,  const int* __restrict__ tgt_c,
    const float* __restrict__ item_emb, const float* __restrict__ cate_emb,
    const bf16* __restrict__ wbf, const float* __restrict__ ab0,
    float* __restrict__ h0, float* __restrict__ sum0, float* __restrict__ sq0)
{
  __shared__ bf16 sX[2][16][520];           // 520-half row pitch: bank-conflict pad
  const int lane = threadIdx.x & 31;
  const int wv   = threadIdx.x >> 5;
  const int nlo  = lane & 15;
  const int koff = (lane >> 4) * 8;
  const int d0   = lane * 4;                // 4 dims of the 128-d q/ub per lane
  const int gw   = blockIdx.x * 2 + wv;
  const int nw   = gridDim.x * 2;
  float sAcc0 = 0.f, qAcc0 = 0.f, sAcc1 = 0.f, qAcc1 = 0.f;

  for (int tile = gw; tile < NTILES; tile += nw) {
    const int r0 = tile << 4;
    // ---- stage x = [q, ub, q-ub, q*ub] as bf16 in per-wave LDS ----
    for (int rr = 0; rr < 16; ++rr) {
      int r = r0 + rr;
      int b = r / LL;
      int iid = hist_i[r], cid = hist_c[r];
      float m = (iid > 0) ? 1.0f : 0.0f;
      const float* qs;
      const float* us;
      if (lane < 16) {                       // lanes 0-15: item half (dims 0-63)
        qs = item_emb + (size_t)tgt_i[b] * DD + d0;
        us = item_emb + (size_t)iid    * DD + d0;
      } else {                               // lanes 16-31: cate half (dims 64-127)
        qs = cate_emb + (size_t)tgt_c[b] * DD + (d0 - DD);
        us = cate_emb + (size_t)cid    * DD + (d0 - DD);
      }
      float4 q4 = *(const float4*)qs;
      float4 u4 = *(const float4*)us;
      float qa[4] = {q4.x, q4.y, q4.z, q4.w};
      float ua[4] = {u4.x*m, u4.y*m, u4.z*m, u4.w*m};
      v4bf vq, vu, vd, vp;
#pragma unroll
      for (int j = 0; j < 4; ++j) {
        vq[j] = f2bf(qa[j]);
        vu[j] = f2bf(ua[j]);
        vd[j] = f2bf(qa[j] - ua[j]);
        vp[j] = f2bf(qa[j] * ua[j]);
      }
      *(v4bf*)&sX[wv][rr][      d0] = vq;
      *(v4bf*)&sX[wv][rr][128 + d0] = vu;
      *(v4bf*)&sX[wv][rr][256 + d0] = vd;
      *(v4bf*)&sX[wv][rr][384 + d0] = vp;
    }
    // per-wave LDS region: only need the wave's own DS stores complete
    asm volatile("s_wait_dscnt 0x0" ::: "memory");

    v8f c0 = {}; v8f c1 = {};
#pragma unroll
    for (int kb = 0; kb < 16; ++kb) {
      // A fragment from LDS: row = lane&15; two contiguous 16B loads
      const bf16* ax = &sX[wv][nlo][kb*32 + koff];
      v8bf alo = *(const v8bf*)ax;
      v8bf ahi = *(const v8bf*)(ax + 16);
      v16bf a = __builtin_shufflevector(alo, ahi, 0,1,2,3,4,5,6,7,8,9,10,11,12,13,14,15);
      // B fragments: preconverted, pre-swizzled, one 32B vector load each
      v16bf b0 = *(const v16bf*)(wbf + ((size_t)(kb*2 + 0)*32 + lane) * 16);
      v16bf b1 = *(const v16bf*)(wbf + ((size_t)(kb*2 + 1)*32 + lane) * 16);
      c0 = __builtin_amdgcn_wmma_f32_16x16x32_bf16(false, a, false, b0, (short)0, c0, false, false);
      c1 = __builtin_amdgcn_wmma_f32_16x16x32_bf16(false, a, false, b1, (short)0, c1, false, false);
    }
    // epilogue: bias, store h0, accumulate batch-norm stats
    float bias0 = ab0[nlo], bias1 = ab0[16 + nlo];
#pragma unroll
    for (int i = 0; i < 8; ++i) {
      int row = r0 + i + (lane >> 4) * 8;     // C layout: lanes 16-31 hold M+8
      float v0 = c0[i] + bias0;
      float v1 = c1[i] + bias1;
      h0[(size_t)row*32 + nlo]      = v0;
      h0[(size_t)row*32 + 16 + nlo] = v1;
      sAcc0 += v0; qAcc0 += v0*v0;
      sAcc1 += v1; qAcc1 += v1*v1;
    }
  }
  // combine lane j with j+16 (same channel), one atomic per channel per wave
  sAcc0 += __shfl_down(sAcc0, 16);
  qAcc0 += __shfl_down(qAcc0, 16);
  sAcc1 += __shfl_down(sAcc1, 16);
  qAcc1 += __shfl_down(qAcc1, 16);
  if (lane < 16) {
    atomicAdd(&sum0[nlo],      sAcc0);
    atomicAdd(&sq0 [nlo],      qAcc0);
    atomicAdd(&sum0[16 + nlo], sAcc1);
    atomicAdd(&sq0 [16 + nlo], qAcc1);
  }
}

// ---------------- K2/K4: batch-norm stats -> mu, rsqrt(var+eps) ----------------
__global__ void din_stats(const float* sum, const float* sq, float* mu, float* rs, int n) {
  int t = threadIdx.x;
  if (t < n) {
    float inv = 1.0f / (float)BL;
    float m = sum[t] * inv;
    float v = sq[t] * inv - m * m;
    mu[t] = m;
    rs[t] = rsqrtf(v + 1e-9f);
  }
}

// ---------------- K3: Dice(h0) + GEMM1 (32->16) + stats ----------------
__global__ __launch_bounds__(64) void din_act1(
    const float* __restrict__ h0, const float* __restrict__ aw1, const float* __restrict__ ab1,
    const float* __restrict__ aal0, const float* __restrict__ mu0, const float* __restrict__ rs0,
    float* __restrict__ h1, float* __restrict__ sum1, float* __restrict__ sq1)
{
  const int lane = threadIdx.x & 31;
  const int nlo  = lane & 15;
  const int koff = (lane >> 4) * 8;
  const int gw   = blockIdx.x * 2 + (threadIdx.x >> 5);
  const int nw   = gridDim.x * 2;
  // B fragment (aw1 is [16 out][32 in]) + per-lane Dice params, hoisted out of loop
  v16bf bfr;
  float muA[16], rsA[16], alA[16];
#pragma unroll
  for (int i = 0; i < 8; ++i) {
    bfr[i]   = f2bf(aw1[nlo*32 + koff + i]);
    bfr[8+i] = f2bf(aw1[nlo*32 + koff + 16 + i]);
    int ca = koff + i, cb = koff + 16 + i;
    muA[i] = mu0[ca]; muA[8+i] = mu0[cb];
    rsA[i] = rs0[ca]; rsA[8+i] = rs0[cb];
    alA[i] = aal0[ca]; alA[8+i] = aal0[cb];
  }
  float bias = ab1[nlo];
  float sAcc = 0.f, qAcc = 0.f;

  for (int tile = gw; tile < NTILES; tile += nw) {
    const int r0 = tile << 4;
    const float* p = h0 + (size_t)(r0 + nlo) * 32 + koff;
    v16bf a;
#pragma unroll
    for (int i = 0; i < 8; ++i) {
      a[i]   = f2bf(dice_f(p[i],      muA[i],   rsA[i],   alA[i]));
      a[8+i] = f2bf(dice_f(p[16 + i], muA[8+i], rsA[8+i], alA[8+i]));
    }
    v8f c = {};
    c = __builtin_amdgcn_wmma_f32_16x16x32_bf16(false, a, false, bfr, (short)0, c, false, false);
#pragma unroll
    for (int i = 0; i < 8; ++i) {
      int row = r0 + i + (lane >> 4) * 8;
      float v = c[i] + bias;
      h1[(size_t)row*16 + nlo] = v;
      sAcc += v; qAcc += v*v;
    }
  }
  sAcc += __shfl_down(sAcc, 16);
  qAcc += __shfl_down(qAcc, 16);
  if (lane < 16) {
    atomicAdd(&sum1[nlo], sAcc);
    atomicAdd(&sq1 [nlo], qAcc);
  }
}

// ---------------- K5: per-batch softmax + interest + final MLP + loss ----------------
__global__ __launch_bounds__(256) void din_final(
    const int* __restrict__ hist_i, const int* __restrict__ hist_c,
    const int* __restrict__ tgt_i,  const int* __restrict__ tgt_c,
    const int* __restrict__ click,
    const float* __restrict__ item_emb, const float* __restrict__ cate_emb,
    const float* __restrict__ aal1, const float* __restrict__ aw2, const float* __restrict__ ab2,
    const float* __restrict__ mw0, const float* __restrict__ mb0,
    const float* __restrict__ mw1, const float* __restrict__ mb1,
    const float* __restrict__ mw2, const float* __restrict__ mb2,
    const float* __restrict__ h1, const float* __restrict__ mu1, const float* __restrict__ rs1,
    float* __restrict__ out)
{
  __shared__ float sScore[LL];
  __shared__ float sAttn[LL];
  __shared__ float sMask[LL];
  __shared__ int   sIid[LL];
  __shared__ int   sCid[LL];
  __shared__ float sZ[256];
  __shared__ float sZ1[128];
  __shared__ float sZ2[64];
  __shared__ float red[256];
  const int b = blockIdx.x;
  const int t = threadIdx.x;

  // scores: Dice(h1) . aw2 + ab2, masked
  float lmax = -3.0e38f;
  if (t < LL) {
    int r = b*LL + t;
    int iid = hist_i[r], cid = hist_c[r];
    float m = (iid > 0) ? 1.0f : 0.0f;
    sIid[t] = iid; sCid[t] = cid; sMask[t] = m;
    float s = ab2[0];
    const float* p = h1 + (size_t)r * 16;
#pragma unroll
    for (int k = 0; k < 16; ++k)
      s += aw2[k] * dice_f(p[k], mu1[k], rs1[k], aal1[k]);
    s -= (1.0f - m) * 1e9f;
    sScore[t] = s;
    lmax = s;
  }
  red[t] = lmax; __syncthreads();
  for (int off = 128; off > 0; off >>= 1) {
    if (t < off) red[t] = fmaxf(red[t], red[t + off]);
    __syncthreads();
  }
  float smax = red[0];
  __syncthreads();
  float lsum = 0.0f;
  if (t < LL) { float e = expf(sScore[t] - smax); sAttn[t] = e; lsum = e; }
  red[t] = lsum; __syncthreads();
  for (int off = 128; off > 0; off >>= 1) {
    if (t < off) red[t] += red[t + off];
    __syncthreads();
  }
  float inv = 1.0f / red[0];
  __syncthreads();

  // z = [interest(128), ti(64), tc(64)]
  if (t < 128) {
    float acc = 0.0f;
    for (int l = 0; l < LL; ++l) {
      float a = sAttn[l] * inv * sMask[l];
      float v = (t < 64) ? item_emb[(size_t)sIid[l]*DD + t]
                         : cate_emb[(size_t)sCid[l]*DD + (t - 64)];
      acc += a * v;
    }
    sZ[t] = acc;
  } else if (t < 192) {
    sZ[t] = item_emb[(size_t)tgt_i[b]*DD + (t - 128)];
  } else {
    sZ[t] = cate_emb[(size_t)tgt_c[b]*DD + (t - 192)];
  }
  __syncthreads();
  if (t < 128) {
    float a = mb0[t];
    const float* w = mw0 + (size_t)t * 256;
    for (int k = 0; k < 256; ++k) a += w[k] * sZ[k];
    sZ1[t] = fmaxf(a, 0.0f);
  }
  __syncthreads();
  if (t < 64) {
    float a = mb1[t];
    const float* w = mw1 + (size_t)t * 128;
    for (int k = 0; k < 128; ++k) a += w[k] * sZ1[k];
    sZ2[t] = fmaxf(a, 0.0f);
  }
  __syncthreads();
  red[t] = (t < 64) ? mw2[t] * sZ2[t] : 0.0f;
  __syncthreads();
  for (int off = 128; off > 0; off >>= 1) {
    if (t < off) red[t] += red[t + off];
    __syncthreads();
  }
  if (t == 0) {
    float lg = red[0] + mb2[0];
    out[1 + b] = 1.0f / (1.0f + expf(-lg));
    float y = (float)click[b];
    float lb = fmaxf(lg, 0.0f) - lg * y + log1pf(expf(-fabsf(lg)));
    atomicAdd(out, lb * (1.0f / (float)BB));
  }
}

extern "C" void kernel_launch(void* const* d_in, const int* in_sizes, int n_in,
                              void* d_out, int out_size, void* d_ws, size_t ws_size,
                              hipStream_t stream) {
  const int*   hist_i   = (const int*)  d_in[0];
  const int*   hist_c   = (const int*)  d_in[1];
  const int*   tgt_i    = (const int*)  d_in[2];
  const int*   tgt_c    = (const int*)  d_in[3];
  const int*   click    = (const int*)  d_in[4];
  const float* item_emb = (const float*)d_in[5];
  const float* cate_emb = (const float*)d_in[6];
  const float* aw0  = (const float*)d_in[7];
  const float* ab0  = (const float*)d_in[8];
  const float* aal0 = (const float*)d_in[9];
  const float* aw1  = (const float*)d_in[10];
  const float* ab1  = (const float*)d_in[11];
  const float* aal1 = (const float*)d_in[12];
  const float* aw2  = (const float*)d_in[13];
  const float* ab2  = (const float*)d_in[14];
  const float* mw0  = (const float*)d_in[15];
  const float* mb0  = (const float*)d_in[16];
  const float* mw1  = (const float*)d_in[17];
  const float* mb1  = (const float*)d_in[18];
  const float* mw2  = (const float*)d_in[19];
  const float* mb2  = (const float*)d_in[20];
  float* out = (float*)d_out;

  float* h0 = (float*)d_ws;                       // [BL,32]  26.2 MB
  float* h1 = h0 + (size_t)BL * 32;               // [BL,16]  13.1 MB
  float* st = h1 + (size_t)BL * 16;               // 192 stat floats
  float* sum0 = st;        float* sq0 = st + 32;
  float* sum1 = st + 64;   float* sq1 = st + 80;
  float* mu0  = st + 96;   float* rs0 = st + 128;
  float* mu1  = st + 160;  float* rs1 = st + 176;
  bf16*  wbf0 = (bf16*)(st + 192);                // 32 frags x 32 lanes x 16 halves = 32 KB

  din_init <<<1,    128, 0, stream>>>(st, out);
  din_cvtw <<<4,    256, 0, stream>>>(aw0, wbf0);
  din_act0 <<<1024, 64,  0, stream>>>(hist_i, hist_c, tgt_i, tgt_c, item_emb, cate_emb,
                                      wbf0, ab0, h0, sum0, sq0);
  din_stats<<<1,    32,  0, stream>>>(sum0, sq0, mu0, rs0, 32);
  din_act1 <<<2048, 64,  0, stream>>>(h0, aw1, ab1, aal0, mu0, rs0, h1, sum1, sq1);
  din_stats<<<1,    32,  0, stream>>>(sum1, sq1, mu1, rs1, 16);
  din_final<<<1024, 256, 0, stream>>>(hist_i, hist_c, tgt_i, tgt_c, click, item_emb, cate_emb,
                                      aal1, aw2, ab2, mw0, mb0, mw1, mb1, mw2, mb2,
                                      h1, mu1, rs1, out);
}
